// GraphTransformer_17471926960236
// MI455X (gfx1250) — matrix operand order repaired and checked by reading, not compiled, not run on hardware
//
#include <hip/hip_runtime.h>

#define DEV_INLINE __device__ __forceinline__

static const int NN   = 20000;           // nodes
static const int NG   = 200;             // graphs
static const int NED  = 320000;          // edges
static const int NE   = 64;              // NUM_EMB
static const int HC   = 128;             // HEADS * NUM_EMB
static const int FFD  = 256;             // 4 * NUM_EMB
static const int NAUG = NN + NG;         // 20200
static const int E2N  = NED + 2 * NN;    // 360000 (edges before self-loops)
static const int EAUG = E2N + NAUG;      // 380200 (all augmented edges)

typedef __attribute__((ext_vector_type(16))) __bf16 v16bf;
typedef __attribute__((ext_vector_type(8)))  float  v8f;

DEV_INLINE __bf16 f2bf(float f) {
  unsigned u = __float_as_uint(f);
  unsigned short h = (unsigned short)((u + 0x7fffu + ((u >> 16) & 1u)) >> 16);
  return __builtin_bit_cast(__bf16, h);
}
DEV_INLINE float bf2f(__bf16 b) {
  unsigned short h = __builtin_bit_cast(unsigned short, b);
  return __uint_as_float(((unsigned)h) << 16);
}
// monotonic float->uint encoding for atomic max on floats
DEV_INLINE unsigned encf(float x) {
  unsigned b = __float_as_uint(x);
  return (b & 0x80000000u) ? ~b : (b | 0x80000000u);
}
DEV_INLINE float decf(unsigned e) {
  unsigned b = (e & 0x80000000u) ? (e & 0x7fffffffu) : ~e;
  return __uint_as_float(b);
}

DEV_INLINE void storev(float* p, float v)  { *p = v; }
DEV_INLINE void storev(__bf16* p, float v) { *p = f2bf(v); }

// ---------------------------------------------------------------------------
// WMMA GEMM: C[M,N] = act(A[M,K] @ B[K,N] + bias). A,B bf16, C fp32 or bf16.
// Block: 256 threads = 8 wave32. Block tile: 128 rows x 16 cols (grid.y = N/16).
// K must be a multiple of 32 (<=256); lda multiple of 8 (16B-aligned rows).
// B panel staged TRANSPOSED in LDS (sBt[16][256]) so per-lane fragment reads
// are 2x16 contiguous bytes (2x ds_load_b128), matching the A side.
// A tile staged in 16-byte uint4 chunks (global_load_b128 -> ds_store_b128).
// act: 0=none, 1=leaky_relu(0.01), 2=relu
// ---------------------------------------------------------------------------
template <typename OutT>
__global__ __launch_bounds__(256) void wmma_gemm(
    const __bf16* __restrict__ A, int lda, const __bf16* __restrict__ B,
    int ldb, const float* __restrict__ bias, OutT* __restrict__ C, int ldc,
    int M, int K, int act) {
  __shared__ __bf16 sBt[16 * 256];  // [n][k], K<=256
  __shared__ __bf16 sA[128 * 32];   // [r][k-chunk of 32]
  const int tid  = threadIdx.x;
  const int lane = tid & 31;
  const int wv   = tid >> 5;        // 8 waves
  const int lo   = lane & 15;
  const int hi   = lane >> 4;
  const int br0  = blockIdx.x * 128;
  const int bc0  = blockIdx.y * 16;

  // stage B panel transposed: sBt[n*256 + k] = B[k, bc0+n]
  for (int t = tid; t < K * 16; t += 256) {
    int k = t >> 4, n = t & 15;
    sBt[n * 256 + k] = B[(size_t)k * ldb + bc0 + n];
  }

  v8f acc = {};
  for (int kk = 0; kk < K; kk += 32) {
    __syncthreads();
    // 128 rows x 32 cols = 512 chunks of 8 bf16 (16B); 2 chunks per thread
    for (int t = tid; t < 512; t += 256) {
      int r  = t >> 2;
      int cc = (t & 3) * 8;
      int gr = br0 + r;
      uint4 val = {0u, 0u, 0u, 0u};
      if (gr < M)
        val = *(const uint4*)(A + (size_t)gr * lda + kk + cc);
      *(uint4*)(&sA[r * 32 + cc]) = val;
    }
    __syncthreads();
    // A fragment (16x32 bf16, ISA 7.12.2): lanes 0-15 K 0..7 & 16..23,
    // lanes 16-31 K 8..15 & 24..31, row = lane%16. 2x16B contiguous reads.
    v16bf av, bv;
    const __bf16* ap = &sA[(wv * 16 + lo) * 32 + hi * 8];
#pragma unroll
    for (int j = 0; j < 8; ++j) {
      av[j]     = ap[j];
      av[j + 8] = ap[16 + j];
    }
    // B fragment (32x16): col = lane%16, j<8 -> K = kk + 8*hi + j, j>=8 -> K = kk + 16 + 8*hi + (j-8).
    // Transposed panel -> 2x16 contiguous bytes per lane.
    const __bf16* bp = &sBt[lo * 256 + kk + hi * 8];
#pragma unroll
    for (int j = 0; j < 8; ++j) {
      bv[j]     = bp[j];
      bv[j + 8] = bp[16 + j];
    }
    acc = __builtin_amdgcn_wmma_f32_16x16x32_bf16(false, av, false, bv,
                                                  (short)0, acc, false, false);
  }
  const int colg = bc0 + lo;
  const float bsv = bias ? bias[colg] : 0.0f;
#pragma unroll
  for (int v = 0; v < 8; ++v) {
    int row = br0 + wv * 16 + hi * 8 + v;   // C/D layout: VGPR v -> M = v + 8*hi
    if (row < M) {
      float x = acc[v] + bsv;
      if (act == 1) x = (x > 0.0f) ? x : 0.01f * x;
      else if (act == 2) x = (x > 0.0f) ? x : 0.0f;
      storev(&C[(size_t)row * ldc + colg], x);
    }
  }
}

// ---------------------------------------------------------------------------
// Elementwise / scatter kernels
// ---------------------------------------------------------------------------
__global__ void k_f32_to_bf16(const float* __restrict__ s, __bf16* __restrict__ d, int n) {
  int i = blockIdx.x * blockDim.x + threadIdx.x;
  if (i < n) d[i] = f2bf(s[i]);
}

__global__ void k_conv_strided(const float* __restrict__ s, int sld,
                               __bf16* __restrict__ d, int dld, int rows, int cols) {
  int i = blockIdx.x * blockDim.x + threadIdx.x;
  if (i >= rows * cols) return;
  int r = i / cols, c = i - r * cols;
  d[(size_t)r * dld + c] = f2bf(s[(size_t)r * sld + c]);
}

__global__ void k_fill_edges(const int* __restrict__ ei, const int* __restrict__ batch,
                             int* __restrict__ asrc, int* __restrict__ adst) {
  int i = blockIdx.x * blockDim.x + threadIdx.x;
  if (i >= EAUG) return;
  int s, d;
  if (i < NED)            { s = ei[i]; d = ei[NED + i]; }
  else if (i < NED + NN)  { int j = i - NED;      s = j;             d = batch[j] + NN; }
  else if (i < E2N)       { int j = i - NED - NN; s = batch[j] + NN; d = j; }
  else                    { int j = i - E2N;      s = j;             d = j; }
  asrc[i] = s; adst[i] = d;
}

__global__ void k_fill_ep(__bf16* __restrict__ aug_e) {   // one-hot rows [E, E+2N)
  int i = blockIdx.x * blockDim.x + threadIdx.x;
  if (i >= 2 * NN * NE) return;
  aug_e[(size_t)NED * NE + i] = f2bf(((i & 63) == 0) ? 1.0f : 0.0f);
}

__global__ void k_edge_feat_scatter(const __bf16* __restrict__ aug_e,
                                    const int* __restrict__ adst,
                                    float* __restrict__ acc, float* __restrict__ cnt) {
  int i = blockIdx.x * blockDim.x + threadIdx.x;
  if (i >= E2N * NE) return;
  int e = i >> 6, c = i & 63;
  int d = adst[e];
  atomicAdd(&acc[(size_t)d * NE + c], bf2f(aug_e[i]));
  if (c == 0) atomicAdd(&cnt[d], 1.0f);
}

__global__ void k_loop_attr(const float* __restrict__ acc, const float* __restrict__ cnt,
                            __bf16* __restrict__ aug_e) {
  int i = blockIdx.x * blockDim.x + threadIdx.x;
  if (i >= NAUG * NE) return;
  int r = i >> 6;
  float cn = cnt[r]; cn = (cn < 1.0f) ? 1.0f : cn;
  aug_e[(size_t)(E2N + r) * NE + (i & 63)] = f2bf(acc[i] / cn);
}

__global__ void k_msg_scatter(const float* __restrict__ o, const __bf16* __restrict__ aug_e,
                              const int* __restrict__ asrc, const int* __restrict__ adst,
                              float* __restrict__ agg) {
  int i = blockIdx.x * blockDim.x + threadIdx.x;
  if (i >= EAUG * NE) return;
  int e = i >> 6, c = i & 63;
  float m = o[(size_t)asrc[e] * NE + c] + bf2f(aug_e[i]);
  m = (m > 0.0f) ? m : 0.0f;
  atomicAdd(&agg[(size_t)adst[e] * NE + c], m + 1e-7f);
}

__global__ void k_add_conv(const float* __restrict__ a, const float* __restrict__ b,
                           __bf16* __restrict__ d, int n) {
  int i = blockIdx.x * blockDim.x + threadIdx.x;
  if (i < n) d[i] = f2bf(a[i] + b[i]);
}

__global__ void k_logits(const float* __restrict__ q, const float* __restrict__ k,
                         const __bf16* __restrict__ eh, const int* __restrict__ asrc,
                         const int* __restrict__ adst, float* __restrict__ logits) {
  int t = blockIdx.x * blockDim.x + threadIdx.x;
  if (t >= EAUG * 2) return;
  int e = t >> 1, h = t & 1;
  const float*  qp = q  + (size_t)adst[e] * HC + h * NE;
  const float*  kp = k  + (size_t)asrc[e] * HC + h * NE;
  const __bf16* ep = eh + (size_t)e * HC + h * NE;
  float acc = 0.0f;
  for (int c = 0; c < NE; ++c) acc += qp[c] * (kp[c] + bf2f(ep[c]));
  logits[t] = acc * 0.125f;   // 1/sqrt(64)
}

__global__ void k_segmax(const float* __restrict__ logits, const int* __restrict__ adst,
                         unsigned* __restrict__ smax) {
  int t = blockIdx.x * blockDim.x + threadIdx.x;
  if (t >= EAUG * 2) return;
  atomicMax(&smax[adst[t >> 1] * 2 + (t & 1)], encf(logits[t]));
}

__global__ void k_expsum(float* __restrict__ logits, const int* __restrict__ adst,
                         const unsigned* __restrict__ smax, float* __restrict__ ssum) {
  int t = blockIdx.x * blockDim.x + threadIdx.x;
  if (t >= EAUG * 2) return;
  int d = adst[t >> 1], h = t & 1;
  float w = __expf(logits[t] - decf(smax[d * 2 + h]));
  logits[t] = w;                              // reuse buffer as exp weights
  atomicAdd(&ssum[d * 2 + h], w);
}

__global__ void k_attn_scatter(const float* __restrict__ vv, const __bf16* __restrict__ eh,
                               const float* __restrict__ expw, const float* __restrict__ ssum,
                               const int* __restrict__ asrc, const int* __restrict__ adst,
                               float* __restrict__ tout) {
  int i = blockIdx.x * blockDim.x + threadIdx.x;
  if (i >= EAUG * HC) return;
  int e = i >> 7, c = i & 127, h = c >> 6;
  int d = adst[e];
  float alpha = expw[e * 2 + h] / (ssum[d * 2 + h] + 1e-16f);
  float val = (vv[(size_t)asrc[e] * HC + c] + bf2f(eh[i])) * alpha;
  atomicAdd(&tout[(size_t)d * HC + c], val);
}

__global__ void k_ln_stats(const float* __restrict__ o, const float* __restrict__ delta,
                           const int* __restrict__ batch, float* __restrict__ gst) {
  int i = blockIdx.x * blockDim.x + threadIdx.x;
  if (i >= NAUG) return;
  int b = (i < NN) ? batch[i] : (i - NN);
  float s = 0.0f, ss = 0.0f;
  for (int c = 0; c < NE; ++c) {
    float x = o[(size_t)i * NE + c] + delta[(size_t)i * NE + c];
    s += x; ss += x * x;
  }
  atomicAdd(&gst[b * 4 + 0], s);
  atomicAdd(&gst[b * 4 + 1], ss);
  atomicAdd(&gst[b * 4 + 2], 1.0f);
}

__global__ void k_ln_apply(float* __restrict__ o, const float* __restrict__ delta,
                           const int* __restrict__ batch, const float* __restrict__ gst,
                           __bf16* __restrict__ ob) {
  int i = blockIdx.x * blockDim.x + threadIdx.x;
  if (i >= NAUG * NE) return;
  int r = i >> 6;
  int b = (r < NN) ? batch[r] : (r - NN);
  float cnt = gst[b * 4 + 2]; cnt = (cnt < 1.0f) ? 1.0f : cnt;
  float norm = cnt * (float)NE;
  float mean = gst[b * 4 + 0] / norm;
  float var  = gst[b * 4 + 1] / norm - mean * mean;
  float y = (o[i] + delta[i] - mean) * rsqrtf(var + 1e-5f);
  o[i] = y;
  ob[i] = f2bf(y);
}

__global__ void k_node_mean_scatter(const float* __restrict__ o, const int* __restrict__ batch,
                                    float* __restrict__ acc, float* __restrict__ cnt) {
  int i = blockIdx.x * blockDim.x + threadIdx.x;
  if (i >= NN * NE) return;
  int r = i >> 6, c = i & 63;
  atomicAdd(&acc[(size_t)batch[r] * NE + c], o[i]);
  if (c == 0) atomicAdd(&cnt[batch[r]], 1.0f);
}

__global__ void k_out_nodes(const float* __restrict__ o, const float* __restrict__ cbuf,
                            const int* __restrict__ batch, float* __restrict__ out) {
  int i = blockIdx.x * blockDim.x + threadIdx.x;
  if (i >= NN * HC) return;
  int r = i >> 7, c = i & 127;
  out[i] = (c < NE) ? o[(size_t)r * NE + c] : cbuf[(size_t)batch[r] * NE + (c - NE)];
}

__global__ void k_out_glob(const float* __restrict__ acc, const float* __restrict__ cnt,
                           const float* __restrict__ o, const float* __restrict__ cbuf,
                           float* __restrict__ out) {
  int i = blockIdx.x * blockDim.x + threadIdx.x;
  if (i >= NG * 192) return;
  int g = i / 192, c = i - g * 192;
  float v;
  if (c < 64)       { float cn = cnt[g]; cn = (cn < 1.0f) ? 1.0f : cn; v = acc[(size_t)g * NE + c] / cn; }
  else if (c < 128) v = o[(size_t)(NN + g) * NE + (c - 64)];
  else              v = cbuf[(size_t)g * NE + (c - 128)];
  out[(size_t)NN * HC + i] = v;
}

// ---------------------------------------------------------------------------
static inline int cdiv(int a, int b) { return (a + b - 1) / b; }

extern "C" void kernel_launch(void* const* d_in, const int* in_sizes, int n_in,
                              void* d_out, int out_size, void* d_ws, size_t ws_size,
                              hipStream_t stream) {
  (void)in_sizes; (void)n_in; (void)out_size; (void)ws_size;
  const float* f_x    = (const float*)d_in[0];
  const float* f_ea   = (const float*)d_in[1];
  const float* f_cond = (const float*)d_in[2];
  const int*   ei     = (const int*)d_in[3];
  const int*   batch  = (const int*)d_in[4];
  const float* x2hW[3] = {(const float*)d_in[5], (const float*)d_in[7], (const float*)d_in[9]};
  const float* x2hB[3] = {(const float*)d_in[6], (const float*)d_in[8], (const float*)d_in[10]};
  const float* e2hW[3] = {(const float*)d_in[11], (const float*)d_in[13], (const float*)d_in[15]};
  const float* e2hB[3] = {(const float*)d_in[12], (const float*)d_in[14], (const float*)d_in[16]};
  const float* c2hW[3] = {(const float*)d_in[17], (const float*)d_in[19], (const float*)d_in[21]};
  const float* c2hB[3] = {(const float*)d_in[18], (const float*)d_in[20], (const float*)d_in[22]};
  const float* genW = (const float*)d_in[23];
  const float* Wq = (const float*)d_in[24]; const float* bq = (const float*)d_in[25];
  const float* Wk = (const float*)d_in[26]; const float* bk = (const float*)d_in[27];
  const float* Wv = (const float*)d_in[28]; const float* bv = (const float*)d_in[29];
  const float* Wep = (const float*)d_in[30];
  const float* Wsp = (const float*)d_in[31]; const float* bsp = (const float*)d_in[32];
  const float* linW = (const float*)d_in[33]; const float* linB = (const float*)d_in[34];
  const float* ff1W = (const float*)d_in[35]; const float* ff1B = (const float*)d_in[36];
  const float* ff2W = (const float*)d_in[37]; const float* ff2B = (const float*)d_in[38];
  float* out = (float*)d_out;

  // ---- workspace arena ----
  char* base = (char*)d_ws; size_t cur = 0;
  auto alloc = [&](size_t n) -> void* {
    cur = (cur + 255) & ~(size_t)255; void* p = base + cur; cur += n; return p;
  };
  __bf16* wb_x2h[3] = {(__bf16*)alloc(32*64*2), (__bf16*)alloc(64*64*2), (__bf16*)alloc(64*64*2)};
  __bf16* wb_e2h1 = (__bf16*)alloc(32*64*2);   // zero-padded K 16->32
  __bf16* wb_e2h[2] = {(__bf16*)alloc(64*64*2), (__bf16*)alloc(64*64*2)};
  __bf16* wb_c2h[3] = {(__bf16*)alloc(32*64*2), (__bf16*)alloc(64*64*2), (__bf16*)alloc(64*64*2)};
  __bf16* wb_gen = (__bf16*)alloc((size_t)3*64*64*2);
  __bf16* wb_q   = (__bf16*)alloc((size_t)3*128*128*2);
  __bf16* wb_k   = (__bf16*)alloc((size_t)3*128*128*2);
  __bf16* wb_v   = (__bf16*)alloc((size_t)3*128*128*2);
  __bf16* wb_s   = (__bf16*)alloc((size_t)3*128*128*2);
  __bf16* wb_e   = (__bf16*)alloc((size_t)3*64*128*2);
  __bf16* wb_lin = (__bf16*)alloc((size_t)3*128*64*2);
  __bf16* wb_f1  = (__bf16*)alloc((size_t)3*64*256*2);
  __bf16* wb_f2  = (__bf16*)alloc((size_t)3*256*64*2);
  __bf16* xbf  = (__bf16*)alloc((size_t)NN*32*2);
  __bf16* eabf = (__bf16*)alloc((size_t)NED*32*2);  // zero-padded 16 -> 32 cols
  __bf16* cbf  = (__bf16*)alloc((size_t)NG*32*2);
  int* asrc = (int*)alloc((size_t)EAUG*4);
  int* adst = (int*)alloc((size_t)EAUG*4);
  __bf16* aug_e = (__bf16*)alloc((size_t)EAUG*NE*2);
  float* accA = (float*)alloc((size_t)NAUG*NE*4);
  float* cnt  = (float*)alloc((size_t)NAUG*4);
  float* o    = (float*)alloc((size_t)NAUG*NE*4);
  __bf16* ob  = (__bf16*)alloc((size_t)NAUG*NE*2);
  __bf16* aggb= (__bf16*)alloc((size_t)NAUG*NE*2);
  float* cbuf = (float*)alloc((size_t)NG*NE*4);
  __bf16* xin = (__bf16*)alloc((size_t)NAUG*HC*2);
  float* qb = (float*)alloc((size_t)NAUG*HC*4);
  float* kb = (float*)alloc((size_t)NAUG*HC*4);
  float* vb = (float*)alloc((size_t)NAUG*HC*4);
  float* tout = (float*)alloc((size_t)NAUG*HC*4);
  __bf16* toutb = (__bf16*)alloc((size_t)NAUG*HC*2);
  float* delta = (float*)alloc((size_t)NAUG*NE*4);
  __bf16* ffh = (__bf16*)alloc((size_t)NAUG*FFD*2);
  float* logits = (float*)alloc((size_t)EAUG*2*4);
  unsigned* smax = (unsigned*)alloc((size_t)NAUG*2*4);
  float* ssum = (float*)alloc((size_t)NAUG*2*4);
  float* gst = (float*)alloc((size_t)NG*4*4);
  // big region: eh (EAUG x 128 bf16) aliases the MLP ping-pong buffers
  char* bigreg = (char*)alloc((size_t)EAUG*HC*2);
  __bf16* eh   = (__bf16*)bigreg;
  __bf16* tmp1 = (__bf16*)bigreg;
  __bf16* tmp2 = (__bf16*)(bigreg + (size_t)NED*NE*2);

  auto conv = [&](const float* s, __bf16* d, int n) {
    k_f32_to_bf16<<<cdiv(n, 256), 256, 0, stream>>>(s, d, n);
  };
  auto gemmF = [&](const __bf16* A, int lda, const __bf16* B, int N,
                   const float* bias, float* C, int ldc, int M, int K, int act) {
    dim3 g(cdiv(M, 128), N / 16);
    wmma_gemm<float><<<g, 256, 0, stream>>>(A, lda, B, N, bias, C, ldc, M, K, act);
  };
  auto gemmB = [&](const __bf16* A, int lda, const __bf16* B, int N,
                   const float* bias, __bf16* C, int ldc, int M, int K, int act) {
    dim3 g(cdiv(M, 128), N / 16);
    wmma_gemm<__bf16><<<g, 256, 0, stream>>>(A, lda, B, N, bias, C, ldc, M, K, act);
  };

  // ---- weight + input conversion to bf16 (K=16 operands zero-padded to 32) ----
  conv(x2hW[0], wb_x2h[0], 32*64); conv(x2hW[1], wb_x2h[1], 64*64); conv(x2hW[2], wb_x2h[2], 64*64);
  hipMemsetAsync(wb_e2h1, 0, (size_t)32*64*2, stream);
  conv(e2hW[0], wb_e2h1, 16*64);                 // rows 0..15 valid, 16..31 zero
  conv(e2hW[1], wb_e2h[0], 64*64); conv(e2hW[2], wb_e2h[1], 64*64);
  conv(c2hW[0], wb_c2h[0], 32*64); conv(c2hW[1], wb_c2h[1], 64*64); conv(c2hW[2], wb_c2h[2], 64*64);
  conv(genW, wb_gen, 3*64*64);
  conv(Wq, wb_q, 3*128*128); conv(Wk, wb_k, 3*128*128);
  conv(Wv, wb_v, 3*128*128); conv(Wsp, wb_s, 3*128*128);
  conv(Wep, wb_e, 3*64*128); conv(linW, wb_lin, 3*128*64);
  conv(ff1W, wb_f1, 3*64*256); conv(ff2W, wb_f2, 3*256*64);
  conv(f_x, xbf, NN*32); conv(f_cond, cbf, NG*32);
  hipMemsetAsync(eabf, 0, (size_t)NED*32*2, stream);
  k_conv_strided<<<cdiv(NED*16, 256), 256, 0, stream>>>(f_ea, 16, eabf, 32, NED, 16);

  // ---- input MLPs (leaky, leaky, none) ----
  gemmB(xbf, 32, wb_x2h[0], 64, x2hB[0], tmp1, 64, NN, 32, 1);
  gemmB(tmp1, 64, wb_x2h[1], 64, x2hB[1], tmp2, 64, NN, 64, 1);
  gemmF(tmp2, 64, wb_x2h[2], 64, x2hB[2], o, 64, NN, 64, 0);
  gemmB(cbf, 32, wb_c2h[0], 64, c2hB[0], tmp1, 64, NG, 32, 1);
  gemmB(tmp1, 64, wb_c2h[1], 64, c2hB[1], tmp2, 64, NG, 64, 1);
  gemmF(tmp2, 64, wb_c2h[2], 64, c2hB[2], cbuf, 64, NG, 64, 0);
  hipMemcpyAsync(o + (size_t)NN * NE, cbuf, (size_t)NG * NE * 4,
                 hipMemcpyDeviceToDevice, stream);
  gemmB(eabf, 32, wb_e2h1, 64, e2hB[0], tmp1, 64, NED, 32, 1);
  gemmB(tmp1, 64, wb_e2h[0], 64, e2hB[1], tmp2, 64, NED, 64, 1);
  gemmB(tmp2, 64, wb_e2h[1], 64, e2hB[2], aug_e, 64, NED, 64, 0);

  // ---- augmented edges, one-hot pads, self-loop attrs (seg-mean) ----
  k_fill_edges<<<cdiv(EAUG, 256), 256, 0, stream>>>(ei, batch, asrc, adst);
  k_fill_ep<<<cdiv(2 * NN * NE, 256), 256, 0, stream>>>(aug_e);
  hipMemsetAsync(accA, 0, (size_t)NAUG * NE * 4, stream);
  hipMemsetAsync(cnt, 0, (size_t)NAUG * 4, stream);
  k_edge_feat_scatter<<<cdiv(E2N * NE, 256), 256, 0, stream>>>(aug_e, adst, accA, cnt);
  k_loop_attr<<<cdiv(NAUG * NE, 256), 256, 0, stream>>>(accA, cnt, aug_e);

  // ---- transformer layers ----
  for (int L = 0; L < 3; ++L) {
    const __bf16* genW_i = wb_gen + (size_t)L * 64 * 64;
    const __bf16* wq_i = wb_q + (size_t)L * 128 * 128;
    const __bf16* wk_i = wb_k + (size_t)L * 128 * 128;
    const __bf16* wv_i = wb_v + (size_t)L * 128 * 128;
    const __bf16* ws_i = wb_s + (size_t)L * 128 * 128;
    const __bf16* we_i = wb_e + (size_t)L * 64 * 128;
    const __bf16* wl_i = wb_lin + (size_t)L * 128 * 64;
    const __bf16* w1_i = wb_f1 + (size_t)L * 64 * 256;
    const __bf16* w2_i = wb_f2 + (size_t)L * 256 * 64;
    const float* bq_i = bq + (size_t)L * HC;
    const float* bk_i = bk + (size_t)L * HC;
    const float* bv_i = bv + (size_t)L * HC;
    const float* bs_i = bsp + (size_t)L * HC;
    const float* bl_i = linB + (size_t)L * NE;
    const float* b1_i = ff1B + (size_t)L * FFD;
    const float* b2_i = ff2B + (size_t)L * NE;

    // generalized message passing -> agg
    hipMemsetAsync(accA, 0, (size_t)NAUG * NE * 4, stream);
    k_msg_scatter<<<cdiv(EAUG * NE, 256), 256, 0, stream>>>(o, aug_e, asrc, adst, accA);
    k_add_conv<<<cdiv(NAUG * NE, 256), 256, 0, stream>>>(accA, o, aggb, NAUG * NE);
    gemmB(aggb, 64, genW_i, 64, nullptr, xin + 64, 128, NAUG, 64, 0);  // xin[:,64:]
    k_conv_strided<<<cdiv(NAUG * NE, 256), 256, 0, stream>>>(o, 64, xin, 128, NAUG, 64);

    // projections
    gemmF(xin, 128, wq_i, 128, bq_i, qb, 128, NAUG, 128, 0);
    gemmF(xin, 128, wk_i, 128, bk_i, kb, 128, NAUG, 128, 0);
    gemmF(xin, 128, wv_i, 128, bv_i, vb, 128, NAUG, 128, 0);
    gemmF(xin, 128, ws_i, 128, bs_i, tout, 128, NAUG, 128, 0);  // skip term, scatter adds on top
    gemmB(aug_e, 64, we_i, 128, nullptr, eh, 128, EAUG, 64, 0);

    // segment softmax attention
    k_logits<<<cdiv(EAUG * 2, 256), 256, 0, stream>>>(qb, kb, eh, asrc, adst, logits);
    hipMemsetAsync(smax, 0, (size_t)NAUG * 2 * 4, stream);
    k_segmax<<<cdiv(EAUG * 2, 256), 256, 0, stream>>>(logits, adst, smax);
    hipMemsetAsync(ssum, 0, (size_t)NAUG * 2 * 4, stream);
    k_expsum<<<cdiv(EAUG * 2, 256), 256, 0, stream>>>(logits, adst, smax, ssum);
    k_attn_scatter<<<cdiv(EAUG * HC, 256), 256, 0, stream>>>(vb, eh, logits, ssum,
                                                             asrc, adst, tout);

    // lin projection + graph layernorm
    k_f32_to_bf16<<<cdiv(NAUG * HC, 256), 256, 0, stream>>>(tout, toutb, NAUG * HC);
    gemmF(toutb, 128, wl_i, 64, bl_i, delta, 64, NAUG, 128, 0);
    hipMemsetAsync(gst, 0, (size_t)NG * 4 * 4, stream);
    k_ln_stats<<<cdiv(NAUG, 256), 256, 0, stream>>>(o, delta, batch, gst);
    k_ln_apply<<<cdiv(NAUG * NE, 256), 256, 0, stream>>>(o, delta, batch, gst, ob);

    // feed-forward + graph layernorm
    gemmB(ob, 64, w1_i, 256, b1_i, ffh, 256, NAUG, 64, 1);
    gemmF(ffh, 256, w2_i, 64, b2_i, delta, 64, NAUG, 256, 0);
    hipMemsetAsync(gst, 0, (size_t)NG * 4 * 4, stream);
    k_ln_stats<<<cdiv(NAUG, 256), 256, 0, stream>>>(o, delta, batch, gst);
    k_ln_apply<<<cdiv(NAUG * NE, 256), 256, 0, stream>>>(o, delta, batch, gst, ob);
  }

  // ---- outputs ----
  hipMemsetAsync(accA, 0, (size_t)NAUG * NE * 4, stream);
  hipMemsetAsync(cnt, 0, (size_t)NAUG * 4, stream);
  k_node_mean_scatter<<<cdiv(NN * NE, 256), 256, 0, stream>>>(o, batch, accA, cnt);
  k_out_nodes<<<cdiv(NN * HC, 256), 256, 0, stream>>>(o, cbuf, batch, out);
  k_out_glob<<<cdiv(NG * 192, 256), 256, 0, stream>>>(accA, cnt, o, cbuf, out);
}
